// MyLSTM_21964462752206
// MI455X (gfx1250) — matrix operand order repaired and checked by reading, not compiled, not run on hardware
//
#include <hip/hip_runtime.h>
#include <hip/hip_bf16.h>

// ---------------------------------------------------------------------------
// LSTM  B=64  S=512  I=1024  H=1024   (gates G = 4H = 4096)
// Phase 1: xW[b,s,g] = sum_i x[b,s,i]*W[g,i] + bias[g]   (bf16 WMMA, f32 acc,
//          ping-pong double-buffered, 2x4 tiles per wave, NT stores)
// Phase 2: ONE persistent kernel, 512 steps with device-scope grid barrier.
//          pre = xW[:,s,:] + h @ Wh^T ; c kept in registers for all steps.
//          Wh stays L2-resident; xW streamed with NT loads.
// ---------------------------------------------------------------------------

#define LSTM_B 64
#define LSTM_S 512
#define LSTM_I 1024
#define LSTM_H 1024
#define LSTM_G 4096   // 4*H
#define LSTM_WK 2048  // W row length (I + H)
#define NBLK_STEP 32  // persistent kernel blocks (256 waves)

typedef __attribute__((ext_vector_type(16))) __bf16 v16bf;
typedef __attribute__((ext_vector_type(8)))  __bf16 v8bf;
typedef __attribute__((ext_vector_type(8)))  float  v8f;

__device__ __forceinline__ float sigmoid_f(float x) {
    return 1.0f / (1.0f + __expf(-x));
}
__device__ __forceinline__ float tanh_f(float x) {
    return 2.0f / (1.0f + __expf(-2.0f * x)) - 1.0f;
}

// A fragment: 16 contiguous K values (32B, 32B-aligned by construction).
__device__ __forceinline__ v16bf make_afrag(const __bf16* p) {
    return *(const v16bf*)p;
}
// B fragment: 16 contiguous K values (32B, 32B-aligned by construction).
__device__ __forceinline__ v16bf load_bfrag(const __bf16* p) {
    return *(const v16bf*)p;
}

__device__ __forceinline__ v8f wmma_bf16(v16bf a, v16bf b, v8f c) {
    return __builtin_amdgcn_wmma_f32_16x16x32_bf16(false, a, false, b,
                                                   (short)0, c, false, false);
}

// ---------------------------------------------------------------------------
// fp32 -> bf16 conversion (grid-stride; src is streamed once -> NT loads)
// ---------------------------------------------------------------------------
__global__ __launch_bounds__(256) void cvt_f32_bf16_kernel(const float* __restrict__ src,
                                                           __bf16* __restrict__ dst,
                                                           long long n) {
    long long i = (long long)blockIdx.x * blockDim.x + threadIdx.x;
    long long stride = (long long)gridDim.x * blockDim.x;
    for (; i < n; i += stride)
        dst[i] = (__bf16)__builtin_nontemporal_load(&src[i]);
}

// Zero h0 (bf16, B*H elements) and the grid-barrier counter.
__global__ __launch_bounds__(256) void init_state_kernel(__bf16* __restrict__ h0,
                                                         unsigned* __restrict__ barrier_ct) {
    int i = blockIdx.x * blockDim.x + threadIdx.x;  // grid sized exactly B*H
    h0[i] = (__bf16)0.0f;
    if (i == 0) *barrier_ct = 0u;
}

// ---------------------------------------------------------------------------
// Phase 1:  xW[row, n] = sum_k xbf[row, k] * Wbf[n, k] + bias[n]
//   rows = B*S = 32768, n in [0,4096), k in [0,1024)
//   one wave -> 32 rows x 64 cols (2 M-tiles x 4 N-tiles, shared fragments),
//   ping-pong double-buffered (no fragment copies -> no WAR hazard nops).
//   grid: 1024 M-groups * 64 N-groups = 65536 waves = 8192 blocks of 8 waves
// ---------------------------------------------------------------------------
__global__ __launch_bounds__(256) void xw_gemm_kernel(const __bf16* __restrict__ xbf,
                                                      const __bf16* __restrict__ Wbf,
                                                      const float* __restrict__ bias,
                                                      float* __restrict__ xW) {
    const int lane = threadIdx.x & 31;
    const int wave = blockIdx.x * (blockDim.x >> 5) + (threadIdx.x >> 5);
    const int nt = wave & 63;   // 64 groups of 64 columns
    const int mg = wave >> 6;   // 1024 groups of 32 rows
    const int m0 = mg * 32;
    const int n0 = nt * 64;

    const int kAoff = (lane & 16) ? 16 : 0;
    const int kBoff = (lane & 16) ? 16 : 0;
    const __bf16* arow0 = xbf + (long long)(m0 + (lane & 15)) * LSTM_I + kAoff;
    const __bf16* arow1 = arow0 + 16 * LSTM_I;
    const __bf16* brow[4];
#pragma unroll
    for (int t = 0; t < 4; ++t)
        brow[t] = Wbf + (long long)(n0 + t * 16 + (lane & 15)) * LSTM_WK + kBoff;

    v8f acc[2][4];
#pragma unroll
    for (int m = 0; m < 2; ++m)
#pragma unroll
        for (int t = 0; t < 4; ++t) acc[m][t] = (v8f){};

    // ping-pong fragment buffers: stage p holds k0 + 32*p
    v16bf a_f[2][2], b_f[2][4];
    a_f[0][0] = make_afrag(arow0);
    a_f[0][1] = make_afrag(arow1);
#pragma unroll
    for (int t = 0; t < 4; ++t) b_f[0][t] = load_bfrag(brow[t]);

    for (int k0 = 0; k0 < LSTM_I - 64; k0 += 64) {
        // prefetch stage 1 (k0+32)
        a_f[1][0] = make_afrag(arow0 + k0 + 32);
        a_f[1][1] = make_afrag(arow1 + k0 + 32);
#pragma unroll
        for (int t = 0; t < 4; ++t) b_f[1][t] = load_bfrag(brow[t] + k0 + 32);
        // compute stage 0 (k0)
#pragma unroll
        for (int m = 0; m < 2; ++m)
#pragma unroll
            for (int t = 0; t < 4; ++t)
                acc[m][t] = wmma_bf16(a_f[0][m], b_f[0][t], acc[m][t]);
        // prefetch stage 0 (k0+64)
        a_f[0][0] = make_afrag(arow0 + k0 + 64);
        a_f[0][1] = make_afrag(arow1 + k0 + 64);
#pragma unroll
        for (int t = 0; t < 4; ++t) b_f[0][t] = load_bfrag(brow[t] + k0 + 64);
        // compute stage 1 (k0+32)
#pragma unroll
        for (int m = 0; m < 2; ++m)
#pragma unroll
            for (int t = 0; t < 4; ++t)
                acc[m][t] = wmma_bf16(a_f[1][m], b_f[1][t], acc[m][t]);
    }
    {   // tail: k0 = I-64 ; stages I-64 and I-32
        const int k0 = LSTM_I - 64;
        a_f[1][0] = make_afrag(arow0 + k0 + 32);
        a_f[1][1] = make_afrag(arow1 + k0 + 32);
#pragma unroll
        for (int t = 0; t < 4; ++t) b_f[1][t] = load_bfrag(brow[t] + k0 + 32);
#pragma unroll
        for (int m = 0; m < 2; ++m)
#pragma unroll
            for (int t = 0; t < 4; ++t)
                acc[m][t] = wmma_bf16(a_f[0][m], b_f[0][t], acc[m][t]);
#pragma unroll
        for (int m = 0; m < 2; ++m)
#pragma unroll
            for (int t = 0; t < 4; ++t)
                acc[m][t] = wmma_bf16(a_f[1][m], b_f[1][t], acc[m][t]);
    }

    // Store with bias; xW is a 512 MiB single-use stream -> non-temporal.
#pragma unroll
    for (int m = 0; m < 2; ++m) {
        const int rbase = m0 + m * 16 + ((lane & 16) ? 8 : 0);
#pragma unroll
        for (int r = 0; r < 8; ++r) {
            long long row = rbase + r;
#pragma unroll
            for (int t = 0; t < 4; ++t) {
                int n = n0 + t * 16 + (lane & 15);
                __builtin_nontemporal_store(acc[m][t][r] + bias[n],
                                            &xW[row * LSTM_G + n]);
            }
        }
    }
}

// ---------------------------------------------------------------------------
// Device-scope grid barrier for the persistent recurrence kernel.
//   release-fence -> syncthreads -> (t0) relaxed add + monotonic spin ->
//   syncthreads -> acquire-fence (invalidates WGP-local caches).
// ---------------------------------------------------------------------------
__device__ __forceinline__ void grid_barrier(unsigned* bar, unsigned target) {
    __builtin_amdgcn_fence(__ATOMIC_RELEASE, "agent");
    __syncthreads();
    if (threadIdx.x == 0) {
        __hip_atomic_fetch_add(bar, 1u, __ATOMIC_RELAXED, __HIP_MEMORY_SCOPE_AGENT);
        while (__hip_atomic_load(bar, __ATOMIC_RELAXED, __HIP_MEMORY_SCOPE_AGENT) < target)
            __builtin_amdgcn_s_sleep(1);
    }
    __syncthreads();
    __builtin_amdgcn_fence(__ATOMIC_ACQUIRE, "agent");
}

// ---------------------------------------------------------------------------
// Phase 2: persistent kernel, all 512 steps.
//   Each wave owns a 16(batch) x 16(hidden) block for ALL FOUR gates;
//   the cell state c stays in registers for the whole sequence.
//   pre[b,g] = xW[b,s,g] + sum_k h_in[b,k] * Wh[g,k],  Wh[g,k] = Wbf[g, I+k].
//   grid: exactly NBLK_STEP blocks * 8 waves = 256 waves (co-resident).
// ---------------------------------------------------------------------------
__global__ __launch_bounds__(256) void lstm_persistent_kernel(
    const float* __restrict__ xW,
    const __bf16* __restrict__ Wbf,
    __bf16* __restrict__ hbuf0,   // zero-initialized (h at s=0)
    __bf16* __restrict__ hbuf1,
    float* __restrict__ out_seq,
    float* __restrict__ hn_out,
    float* __restrict__ cn_out,
    unsigned* __restrict__ barrier_ct) {
    const int lane = threadIdx.x & 31;
    const int wave = blockIdx.x * (blockDim.x >> 5) + (threadIdx.x >> 5);
    const int mt = wave & 3;    // 4 batch tiles (64/16)
    const int nt = wave >> 2;   // 64 hidden tiles (1024/16)
    const int m0 = mt * 16;
    const int j0 = nt * 16;

    const int kAoff = (lane & 16) ? 16 : 0;
    const int kBoff = (lane & 16) ? 16 : 0;
    const int arow_off = (m0 + (lane & 15)) * LSTM_H + kAoff;
    const __bf16* brow[4];
#pragma unroll
    for (int g = 0; g < 4; ++g)
        brow[g] = Wbf + (long long)(g * LSTM_H + j0 + (lane & 15)) * LSTM_WK +
                  LSTM_I + kBoff;

    const int rbase = m0 + ((lane & 16) ? 8 : 0);
    const int j = j0 + (lane & 15);

    v8f creg = (v8f){};  // cell state tile, lives in registers across all steps

    for (int s = 0; s < LSTM_S; ++s) {
        const __bf16* h_in = (s & 1) ? hbuf1 : hbuf0;
        __bf16* h_out = (s & 1) ? hbuf0 : hbuf1;

        // Issue the xW loads first: independent of h, latency hides under GEMM.
        // Single-use stream -> non-temporal loads (keep Wh resident in L2).
        float xwv[4][8];
#pragma unroll
        for (int r = 0; r < 8; ++r) {
            const long long xrow = ((long long)(rbase + r) * LSTM_S + s) * LSTM_G;
#pragma unroll
            for (int g = 0; g < 4; ++g)
                xwv[g][r] = __builtin_nontemporal_load(&xW[xrow + g * LSTM_H + j]);
        }

        // GEMM: h @ Wh^T, ping-pong double-buffered (no fragment copies).
        const __bf16* arow = h_in + arow_off;
        v8f acc[4];
#pragma unroll
        for (int g = 0; g < 4; ++g) acc[g] = (v8f){};

        v16bf a_f[2], b_f[2][4];
        a_f[0] = make_afrag(arow);
#pragma unroll
        for (int g = 0; g < 4; ++g) b_f[0][g] = load_bfrag(brow[g]);

        for (int k0 = 0; k0 < LSTM_H - 64; k0 += 64) {
            a_f[1] = make_afrag(arow + k0 + 32);
#pragma unroll
            for (int g = 0; g < 4; ++g) b_f[1][g] = load_bfrag(brow[g] + k0 + 32);
#pragma unroll
            for (int g = 0; g < 4; ++g) acc[g] = wmma_bf16(a_f[0], b_f[0][g], acc[g]);
            a_f[0] = make_afrag(arow + k0 + 64);
#pragma unroll
            for (int g = 0; g < 4; ++g) b_f[0][g] = load_bfrag(brow[g] + k0 + 64);
#pragma unroll
            for (int g = 0; g < 4; ++g) acc[g] = wmma_bf16(a_f[1], b_f[1][g], acc[g]);
        }
        {   // tail: k0 = H-64 ; stages H-64 and H-32
            const int k0 = LSTM_H - 64;
            a_f[1] = make_afrag(arow + k0 + 32);
#pragma unroll
            for (int g = 0; g < 4; ++g) b_f[1][g] = load_bfrag(brow[g] + k0 + 32);
#pragma unroll
            for (int g = 0; g < 4; ++g) acc[g] = wmma_bf16(a_f[0], b_f[0][g], acc[g]);
#pragma unroll
            for (int g = 0; g < 4; ++g) acc[g] = wmma_bf16(a_f[1], b_f[1][g], acc[g]);
        }

        // Gates + state update, all in registers.
#pragma unroll
        for (int r = 0; r < 8; ++r) {
            const int bidx = rbase + r;
            float cand = tanh_f(acc[0][r] + xwv[0][r]);
            float fg = sigmoid_f(acc[1][r] + xwv[1][r]);
            float ug = sigmoid_f(acc[2][r] + xwv[2][r]);
            float og = sigmoid_f(acc[3][r] + xwv[3][r]);

            float cnew = ug * cand + fg * creg[r];
            creg[r] = cnew;
            float h = og * tanh_f(cnew);

            __builtin_nontemporal_store(
                h, &out_seq[((long long)bidx * LSTM_S + s) * LSTM_H + j]);
            h_out[bidx * LSTM_H + j] = (__bf16)h;
            if (s == LSTM_S - 1) {
                hn_out[bidx * LSTM_H + j] = h;
                cn_out[bidx * LSTM_H + j] = cnew;
            }
        }

        grid_barrier(barrier_ct, (unsigned)NBLK_STEP * (unsigned)(s + 1));
    }
}

// ---------------------------------------------------------------------------
extern "C" void kernel_launch(void* const* d_in, const int* in_sizes, int n_in,
                              void* d_out, int out_size, void* d_ws, size_t ws_size,
                              hipStream_t stream) {
    const float* x = (const float*)d_in[0];   // (B,S,I)
    const float* W = (const float*)d_in[1];   // (4H, I+H)
    const float* b = (const float*)d_in[2];   // (4H,)

    float* out_seq = (float*)d_out;                                   // (B,S,H)
    float* hn = out_seq + (long long)LSTM_B * LSTM_S * LSTM_H;        // (1,B,H)
    float* cn = hn + (long long)LSTM_B * LSTM_H;                      // (1,B,H)

    // workspace layout
    char* ws = (char*)d_ws;
    const long long nW = (long long)LSTM_G * LSTM_WK;           //  8,388,608
    const long long nX = (long long)LSTM_B * LSTM_S * LSTM_I;   // 33,554,432
    const long long nXW = (long long)LSTM_B * LSTM_S * LSTM_G;  // 134,217,728
    const long long nBH = (long long)LSTM_B * LSTM_H;           //     65,536

    __bf16* Wbf = (__bf16*)ws;       ws += nW * 2;    //  16 MiB
    __bf16* xbf = (__bf16*)ws;       ws += nX * 2;    //  64 MiB
    float*  xW  = (float*)ws;        ws += nXW * 4;   // 512 MiB
    __bf16* hA  = (__bf16*)ws;       ws += nBH * 2;
    __bf16* hB  = (__bf16*)ws;       ws += nBH * 2;
    unsigned* barrier_ct = (unsigned*)ws;

    // 0) convert W and x to bf16; zero h0 and the barrier counter
    cvt_f32_bf16_kernel<<<4096, 256, 0, stream>>>(W, Wbf, nW);
    cvt_f32_bf16_kernel<<<8192, 256, 0, stream>>>(x, xbf, nX);
    init_state_kernel<<<(int)(nBH / 256), 256, 0, stream>>>(hA, barrier_ct);

    // 1) big input GEMM:  xW = x @ Wx^T + bias
    xw_gemm_kernel<<<8192, 256, 0, stream>>>(xbf, Wbf, b, xW);

    // 2) all 512 recurrence steps in one persistent kernel (grid barrier/step)
    lstm_persistent_kernel<<<NBLK_STEP, 256, 0, stream>>>(
        xW, Wbf, hA, hB, out_seq, hn, cn, barrier_ct);
}
